// MultiHeadSelfAttention1D_5471788335637
// MI455X (gfx1250) — matrix operand order repaired and checked by reading, not compiled, not run on hardware
//
#include <hip/hip_runtime.h>
#include <hip/hip_bf16.h>

// Problem constants (match reference)
#define kB  2
#define kC  256
#define kN  4096
#define kH  8
#define kHD 32
#define kSCALE 0.17677669529663687f  // 1/sqrt(32)

typedef __attribute__((ext_vector_type(16))) _Float16 v16h;
typedef __attribute__((ext_vector_type(8)))  _Float16 v8h;
typedef __attribute__((ext_vector_type(8)))  float    v8f;

__device__ __forceinline__ v8f wmma16(v16h a, v16h b, v8f c) {
  // D = A(16x32 f16) * B(32x16 f16) + C(16x16 f32)
  return __builtin_amdgcn_wmma_f32_16x16x32_f16(
      /*neg_a=*/false, a, /*neg_b=*/false, b,
      /*c_mod=*/(short)0, c, /*reuse_a=*/false, /*reuse_b=*/false);
}

// CDNA5 async global->LDS copy (GLOBAL_LOAD_ASYNC_TO_LDS_B128, ASYNCcnt-tracked).
// lds_addr: workgroup-relative LDS byte address in a VGPR; gaddr: 64-bit global address.
__device__ __forceinline__ void async_copy_b128(unsigned int lds_addr, unsigned long long gaddr) {
  asm volatile("global_load_async_to_lds_b128 %0, %1, off"
               :: "v"(lds_addr), "v"(gaddr) : "memory");
}
__device__ __forceinline__ void wait_async0() {
  asm volatile("s_wait_asynccnt 0x0" ::: "memory");
}

__device__ __forceinline__ v16h cat8(v8h lo, v8h hi) {
  return __builtin_shufflevector(lo, hi, 0, 1, 2, 3, 4, 5, 6, 7,
                                 8, 9, 10, 11, 12, 13, 14, 15);
}

// ---- Fragment loaders (wave32 WMMA operands, contiguous k = 16*(lane/16) + elem in both A and B) ----

// A fragment 16x32 from row-major f16 [rows x ldm], element (m,k) = p[m*ldm + k].
// Lane l<16: M=l, elems 0..15 -> K=kb..kb+15 (kb=kk0);
// lanes 16..31: M=l-16, kb=kk0+16.  Two aligned 16B loads (b128).
__device__ __forceinline__ v16h load_a_f16(const _Float16* p, int ldm, int row0, int kk0) {
  const int lane = threadIdx.x & 31;
  const int m  = row0 + (lane & 15);
  const int kb = kk0 + ((lane & 16) ? 16 : 0);
  const _Float16* q = p + (size_t)m * ldm + kb;
  const v8h lo = *(const v8h*)(q);
  const v8h hi = *(const v8h*)(q + 8);
  return cat8(lo, hi);
}

// A fragment 16x32 where element (m,k) lives at x[k*ldk + m] (f32 strided, cvt to f16).
__device__ __forceinline__ v16h load_a_xt(const float* x, int ldk, int row0, int kk0) {
  const int lane = threadIdx.x & 31;
  const int m  = row0 + (lane & 15);
  const int kb = kk0 + ((lane & 16) ? 16 : 0);
  const float* q = x + (size_t)kb * ldk + m;
  v16h a;
#pragma unroll
  for (int e = 0; e < 8; ++e) {
    a[e]     = (_Float16)q[(size_t)e * ldk];
    a[e + 8] = (_Float16)q[(size_t)(e + 8) * ldk];
  }
  return a;
}

// B fragment 32x16 where element (k,n) lives at p[n*ld + k] (B = M^T, M row-major f16).
// Per-lane 32 contiguous bytes -> two b128 loads.
__device__ __forceinline__ v16h load_bt_f16(const _Float16* p, int ld, int kk0, int n0) {
  const int lane = threadIdx.x & 31;
  const int n  = n0 + (lane & 15);
  const int kb = kk0 + ((lane & 16) ? 16 : 0);
  const _Float16* q = p + (size_t)n * ld + kb;
  const v8h lo = *(const v8h*)(q);
  const v8h hi = *(const v8h*)(q + 8);
  return cat8(lo, hi);
}

// B fragment 32x16 from a row-major f16 [32 x 32] tile in LDS: element (k,n) = p[k*32 + n].
// Lane l<16: N=n0+l, K=0..15; lanes 16..31: N=n0+l-16, K=16..31.
__device__ __forceinline__ v16h load_b_lds(const _Float16* p, int n0) {
  const int lane = threadIdx.x & 31;
  const int n  = n0 + (lane & 15);
  const int kb = (lane & 16) ? 16 : 0;
  v16h b;
#pragma unroll
  for (int e = 0; e < 16; ++e) b[e] = p[(kb + e) * 32 + n];
  return b;
}

// B fragment 32x16 from row-major f32 [K x ldn] with f32->f16 convert (strided).
__device__ __forceinline__ v16h load_b_f32(const float* w, int ldn, int kk0, int n0) {
  const int lane = threadIdx.x & 31;
  const int n  = n0 + (lane & 15);
  const int kb = kk0 + ((lane & 16) ? 16 : 0);
  const float* q = w + (size_t)kb * ldn + n;
  v16h b;
#pragma unroll
  for (int e = 0; e < 16; ++e) b[e] = (_Float16)q[(size_t)e * ldn];
  return b;
}

// ---------------------------------------------------------------------------
// Kernel 1: QKV projection.  qkv[b,n,j] = sum_c x[b,c,n]*Wqkv[c,j] + bqkv[j]
// Output scattered to Q/K/V as f16 [B,H,N,HD].  One wave = one 16x16 tile.
// ---------------------------------------------------------------------------
__global__ void __launch_bounds__(256) mhsa_qkv_kernel(
    const float* __restrict__ x, const float* __restrict__ Wqkv,
    const float* __restrict__ bqkv,
    _Float16* __restrict__ Q, _Float16* __restrict__ K, _Float16* __restrict__ V) {
  const int TJ = (3 * kC) / 16;             // 48 col tiles
  const int TN = kN / 16;                   // 256 row tiles
  const int wave = blockIdx.x * (blockDim.x >> 5) + (threadIdx.x >> 5);
  const int b   = wave / (TN * TJ);
  const int rem = wave % (TN * TJ);
  const int n0  = (rem / TJ) * 16;
  const int j0  = (rem % TJ) * 16;

  const float* xb = x + (size_t)b * kC * kN;
  v8f acc = {};
#pragma unroll
  for (int k0 = 0; k0 < kC; k0 += 32) {
    v16h a  = load_a_xt(xb, kN, n0, k0);
    v16h bm = load_b_f32(Wqkv, 3 * kC, k0, j0);
    acc = wmma16(a, bm, acc);
  }

  const int lane = threadIdx.x & 31;
  const int col  = j0 + (lane & 15);
  const float bv = bqkv[col];
  const int which = col >> 8;               // 0=Q, 1=K, 2=V (256-aligned tiles)
  const int cl = col & 255;
  const int h  = cl >> 5;
  const int d  = cl & 31;
  _Float16* dst = (which == 0) ? Q : (which == 1) ? K : V;
  const int rbase = n0 + ((lane & 16) ? 8 : 0);
#pragma unroll
  for (int r = 0; r < 8; ++r) {
    const size_t row = (size_t)rbase + r;
    dst[(((size_t)b * kH + h) * kN + row) * kHD + d] = (_Float16)(acc[r] + bv);
  }
}

// ---------------------------------------------------------------------------
// Kernel 2: streaming (flash) attention.  One wave owns 16 queries of one (b,h);
// loops over keys in tiles of 32.  Block = 8 waves = 128 queries.
// Per-wave LDS: 1KB P-staging (16x32 f16) + 2KB V tile (32x32 f16) = 3KB.
// V tiles are fetched with async global->LDS b128 copies overlapped with QK^T.
// ---------------------------------------------------------------------------
__global__ void __launch_bounds__(256) mhsa_attn_kernel(
    const _Float16* __restrict__ Q, const _Float16* __restrict__ K,
    const _Float16* __restrict__ V, _Float16* __restrict__ Aout) {
  __shared__ _Float16 smem[8 * 1536];       // 24 KB: per wave [P 16x32 | V 32x32]

  const int waveInBlk = threadIdx.x >> 5;
  const int lane      = threadIdx.x & 31;
  const int bh = blockIdx.x / (kN / 128);
  const int qc = blockIdx.x % (kN / 128);
  const int b  = bh / kH;
  const int h  = bh % kH;
  const int q0 = qc * 128 + waveInBlk * 16;

  const _Float16* Qb = Q + (size_t)bh * kN * kHD;
  const _Float16* Kb = K + (size_t)bh * kN * kHD;
  const _Float16* Vb = V + (size_t)bh * kN * kHD;

  _Float16* pbuf = smem + waveInBlk * 1536;            // P staging, 16x32
  _Float16* vbuf = pbuf + 512;                         // V tile, 32x32
  const unsigned int lds_v =
      (unsigned int)(uintptr_t)smem + waveInBlk * 3072u + 1024u;

  // Q fragment: 16 queries x full HD=32 (exactly one WMMA depth). Loaded once.
  const v16h aq = load_a_f16(Qb, kHD, q0, 0);

  float m[8], l[8];
  v8f o0 = {}, o1 = {};
#pragma unroll
  for (int r = 0; r < 8; ++r) { m[r] = -3.0e38f; l[r] = 0.0f; }

  for (int kt = 0; kt < kN; kt += 32) {
    // Kick off async V-tile copy (2 KB contiguous) into this wave's LDS region;
    // it overlaps with the QK^T WMMAs and the softmax update below.
    const unsigned long long gv =
        (unsigned long long)(uintptr_t)(Vb + (size_t)kt * kHD);
#pragma unroll
    for (int c = 0; c < 4; ++c)
      async_copy_b128(lds_v + c * 512u + lane * 16u,
                      gv + (unsigned long long)(c * 512 + lane * 16));
    if (kt + 32 < kN)                        // global_prefetch_b8 next K tile
      __builtin_prefetch(Kb + (size_t)(kt + 32) * kHD, 0, 0);

    // S = Q * K^T  (two 16x16 tiles covering 32 keys); K^T frags are b128 loads.
    const v16h bk0 = load_bt_f16(Kb, kHD, 0, kt);
    const v16h bk1 = load_bt_f16(Kb, kHD, 0, kt + 16);
    v8f z = {};
    v8f s0 = wmma16(aq, bk0, z);
    v8f s1 = wmma16(aq, bk1, z);

    // Online softmax in the C-accumulator layout: row M=r(+8) spans 16 lanes.
#pragma unroll
    for (int r = 0; r < 8; ++r) {
      s0[r] *= kSCALE; s1[r] *= kSCALE;
      float mx = fmaxf(s0[r], s1[r]);
      mx = fmaxf(mx, __shfl_xor(mx, 1, 32));
      mx = fmaxf(mx, __shfl_xor(mx, 2, 32));
      mx = fmaxf(mx, __shfl_xor(mx, 4, 32));
      mx = fmaxf(mx, __shfl_xor(mx, 8, 32));
      const float mnew = fmaxf(m[r], mx);
      const float corr = __expf(m[r] - mnew);
      const float p0 = __expf(s0[r] - mnew);
      const float p1 = __expf(s1[r] - mnew);
      float rs = p0 + p1;
      rs += __shfl_xor(rs, 1, 32);
      rs += __shfl_xor(rs, 2, 32);
      rs += __shfl_xor(rs, 4, 32);
      rs += __shfl_xor(rs, 8, 32);
      l[r] = l[r] * corr + rs;
      m[r] = mnew;
      o0[r] *= corr; o1[r] *= corr;
      s0[r] = p0;    s1[r] = p1;
    }

    // Rearrange P from C-layout to A-layout through wave-private LDS.
    const int prow = (lane & 16) ? 8 : 0;
    const int pcol = lane & 15;
#pragma unroll
    for (int r = 0; r < 8; ++r) {
      pbuf[(prow + r) * 32 + pcol]      = (_Float16)s0[r];
      pbuf[(prow + r) * 32 + 16 + pcol] = (_Float16)s1[r];
    }
    const v16h ap = load_a_f16(pbuf, 32, 0, 0);   // ds_load_b128 x2, same wave

    // O += P * V: fence the async copy, then read V B-frags from LDS.
    wait_async0();
    const v16h bv0 = load_b_lds(vbuf, 0);
    const v16h bv1 = load_b_lds(vbuf, 16);
    o0 = wmma16(ap, bv0, o0);
    o1 = wmma16(ap, bv1, o1);
  }

  // Finalize: O /= l, store f16 attn output in [B, N, C] (C = h*HD + d).
  const int rbase = (lane & 16) ? 8 : 0;
  const int d0    = lane & 15;
#pragma unroll
  for (int r = 0; r < 8; ++r) {
    const float inv = 1.0f / l[r];
    const size_t row  = (size_t)q0 + rbase + r;
    const size_t base = ((size_t)b * kN + row) * kC + (size_t)h * kHD;
    Aout[base + d0]      = (_Float16)(o0[r] * inv);
    Aout[base + 16 + d0] = (_Float16)(o1[r] * inv);
  }
}

// ---------------------------------------------------------------------------
// Kernel 3: output projection + transpose.
// out[b,c,n] = sum_k A[b,n,k]*Wproj[k,c] + bproj[c]
// ---------------------------------------------------------------------------
__global__ void __launch_bounds__(256) mhsa_proj_kernel(
    const _Float16* __restrict__ Ain, const float* __restrict__ Wproj,
    const float* __restrict__ bproj, float* __restrict__ out) {
  const int TC = kC / 16;                   // 16 col tiles
  const int TN = kN / 16;                   // 256 row tiles
  const int wave = blockIdx.x * (blockDim.x >> 5) + (threadIdx.x >> 5);
  const int b   = wave / (TN * TC);
  const int rem = wave % (TN * TC);
  const int n0  = (rem / TC) * 16;
  const int c0  = (rem % TC) * 16;

  const _Float16* Ab = Ain + (size_t)b * kN * kC;
  v8f acc = {};
#pragma unroll
  for (int k0 = 0; k0 < kC; k0 += 32) {
    v16h a  = load_a_f16(Ab, kC, n0, k0);   // b128 loads
    v16h bm = load_b_f32(Wproj, kC, k0, c0);
    acc = wmma16(a, bm, acc);
  }

  const int lane = threadIdx.x & 31;
  const int col  = c0 + (lane & 15);
  const float bv = bproj[col];
  const int rbase = n0 + ((lane & 16) ? 8 : 0);
  // 8 consecutive floats along n per lane -> two b128 stores.
  float4 lo4, hi4;
  lo4.x = acc[0] + bv; lo4.y = acc[1] + bv; lo4.z = acc[2] + bv; lo4.w = acc[3] + bv;
  hi4.x = acc[4] + bv; hi4.y = acc[5] + bv; hi4.z = acc[6] + bv; hi4.w = acc[7] + bv;
  float* o = out + ((size_t)b * kC + col) * kN + rbase;
  *(float4*)(o)     = lo4;
  *(float4*)(o + 4) = hi4;
}

// ---------------------------------------------------------------------------
extern "C" void kernel_launch(void* const* d_in, const int* in_sizes, int n_in,
                              void* d_out, int out_size, void* d_ws, size_t ws_size,
                              hipStream_t stream) {
  (void)in_sizes; (void)n_in; (void)out_size; (void)ws_size;
  const float* x     = (const float*)d_in[0];
  const float* Wqkv  = (const float*)d_in[1];
  const float* bqkv  = (const float*)d_in[2];
  const float* Wproj = (const float*)d_in[3];
  const float* bproj = (const float*)d_in[4];
  float* out = (float*)d_out;

  // Workspace layout (f16): Q | K | V | attn_out, each B*H*N*HD = B*N*C elems.
  const size_t elems = (size_t)kB * kN * kC;      // 2,097,152
  _Float16* Q = (_Float16*)d_ws;
  _Float16* K = Q + elems;
  _Float16* V = K + elems;
  _Float16* A = V + elems;                        // total 16 MB

  // QKV: B*(N/16)*(3C/16) = 24576 wave-tiles / 8 waves per block
  mhsa_qkv_kernel<<<24576 / 8, 256, 0, stream>>>(x, Wqkv, bqkv, Q, K, V);
  // Attention: B*H*(N/128) blocks, 8 waves x 16 queries each
  mhsa_attn_kernel<<<kB * kH * (kN / 128), 256, 0, stream>>>(Q, K, V, A);
  // Projection: B*(N/16)*(C/16) = 8192 wave-tiles / 8
  mhsa_proj_kernel<<<8192 / 8, 256, 0, stream>>>(A, Wproj, bproj, out);
}
